// differentiableSuperpixelTokenizer_13752485282515
// MI455X (gfx1250) — compile-verified, matrix operand order checked by reading, not run on hardware
//
#include <hip/hip_runtime.h>

typedef __attribute__((ext_vector_type(16))) _Float16 v16h;
typedef __attribute__((ext_vector_type(8)))  _Float16 v8h;
typedef __attribute__((ext_vector_type(8)))  float    v8f;
typedef __attribute__((ext_vector_type(2)))  float    v2f;
typedef __attribute__((ext_vector_type(4)))  unsigned int u32x4;
typedef __attribute__((ext_vector_type(8)))  int      i32x8;
typedef __attribute__((ext_vector_type(4)))  int      i32x4;

union Frag16 { v16h v; v8h h[2]; };

#if __has_include(<hip/amd_detail/amd_gfx1250_TDM.h>)
#define TDM_SIX_ARG 1
#endif

#define B_   4
#define CIN  3
#define H_   224
#define W_   224
#define HW   (H_ * W_)       // 50176
#define C1   64
#define E_   768
#define S_   256
#define HID  384

// ISA k-mapping for 16-bit A/B fragments of V_WMMA_*_16X16X32_F16:
// lane L: row = L%16, khalf = L/16.  Element e -> K:
//   e in [0,8):  K = 8*khalf + e          (contiguous run of 8)
//   e in [8,16): K = 16 + 8*khalf + (e-8) (contiguous run of 8)
__device__ __forceinline__ int kmap16(int e, int khalf) {
  return (e < 8) ? (8 * khalf + e) : (16 + 8 * khalf + (e - 8));
}

// ---------------- conv1: 3->64, 3x3, SAME, ReLU -> act1 NHWC f16 ----------
__global__ void conv1_kernel(const float* __restrict__ img,
                             const float* __restrict__ w,
                             _Float16* __restrict__ act1) {
  int t = blockIdx.x * blockDim.x + threadIdx.x;       // NHWC flat index
  if (t >= B_ * HW * C1) return;
  int oc = t % C1;
  int x  = (t / C1) % W_;
  int y  = (t / (C1 * W_)) % H_;
  int b  = t / (C1 * HW);
  float s = 0.f;
  #pragma unroll
  for (int c = 0; c < CIN; ++c)
    #pragma unroll
    for (int dy = 0; dy < 3; ++dy) {
      int yy = y + dy - 1;
      if (yy < 0 || yy >= H_) continue;
      #pragma unroll
      for (int dx = 0; dx < 3; ++dx) {
        int xx = x + dx - 1;
        if (xx < 0 || xx >= W_) continue;
        s += img[((b * CIN + c) * H_ + yy) * W_ + xx] *
             w[((oc * CIN + c) * 3 + dy) * 3 + dx];
      }
    }
  act1[t] = (_Float16)fmaxf(s, 0.f);
}

// --------- weight prepack: fp32 OIHW -> f16 fragment order ---------------
// w2F: [tap(9)][kc2(2)][mt(4)][lane(32)][e(16)]
// w3F: [kc2(2)][mt(48)][lane(32)][e(16)]
__global__ void cvt_kernel(const float* __restrict__ w2, const float* __restrict__ w3,
                           _Float16* __restrict__ w2F, _Float16* __restrict__ w3F) {
  int t = blockIdx.x * blockDim.x + threadIdx.x;
  const int N2 = 9 * 2 * 4 * 32 * 16;    // 36864
  const int N3 = 2 * 48 * 32 * 16;       // 49152
  if (t < N2) {
    int e = t & 15, lane = (t >> 4) & 31, mt = (t >> 9) & 3;
    int kc2 = (t >> 11) & 1, tap = t >> 12;
    int khalf = lane >> 4;
    int m = mt * 16 + (lane & 15);
    int c = kc2 * 32 + kmap16(e, khalf);
    int dy = tap / 3, dx = tap % 3;
    w2F[t] = (_Float16)w2[((m * C1 + c) * 3 + dy) * 3 + dx];
  } else if (t < N2 + N3) {
    int u = t - N2;
    int e = u & 15, lane = (u >> 4) & 31, mt = (u >> 9) % 48;
    int kc2 = u / (48 * 512);
    int khalf = lane >> 4;
    int m = mt * 16 + (lane & 15);
    int c = kc2 * 32 + kmap16(e, khalf);
    w3F[u] = (_Float16)w3[m * C1 + c];
  }
}

// ---------------- conv2: 64->64, 3x3 implicit GEMM, LDS-staged ------------
__global__ void __launch_bounds__(256)
conv2_wmma_kernel(const _Float16* __restrict__ act1,
                  const _Float16* __restrict__ w2F,
                  _Float16* __restrict__ act2) {
  __shared__ _Float16 halo[10 * 18 * C1];        // 11520 halves, [yy][xx][c]
  __shared__ _Float16 stage[8][16 * C1];         // per-wave NHWC stage

  int tid = threadIdx.x;
  int bb  = blockIdx.x;
  int tx  = bb % (W_ / 16);
  int ty  = (bb / (W_ / 16)) % (H_ / 8);
  int b   = bb / ((W_ / 16) * (H_ / 8));
  int x0  = tx * 16, y0 = ty * 8;

  for (int i = tid; i < 180 * 8; i += 256) {
    int chunk = i & 7;
    int pos   = i >> 3;
    int xx = pos % 18, yy = pos / 18;
    int y = y0 + yy - 1, x = x0 + xx - 1;
    float4 val = make_float4(0.f, 0.f, 0.f, 0.f);
    if (y >= 0 && y < H_ && x >= 0 && x < W_)
      val = *(const float4*)(act1 + ((long)(b * H_ + y) * W_ + x) * C1 + chunk * 8);
    *(float4*)(&halo[pos * C1 + chunk * 8]) = val;
  }
  __syncthreads();

  int wv    = tid >> 5;
  int lane  = tid & 31;
  int n     = lane & 15;
  int khalf = lane >> 4;

  v8f acc[4] = {};
  for (int tap = 0; tap < 9; ++tap) {
    int dy = tap / 3, dx = tap % 3;
    const _Float16* hsrc = &halo[((wv + dy) * 18 + (n + dx)) * C1 + khalf * 8];
    #pragma unroll
    for (int kc2 = 0; kc2 < 2; ++kc2) {
      Frag16 bf;
      bf.h[0] = *(const v8h*)(hsrc + kc2 * 32);
      bf.h[1] = *(const v8h*)(hsrc + kc2 * 32 + 16);
      const _Float16* wsrc = w2F + (((tap * 2 + kc2) * 4) * 32 + lane) * 16;
      #pragma unroll
      for (int mt = 0; mt < 4; ++mt) {
        v16h af = *(const v16h*)(wsrc + mt * 512);
        acc[mt] = __builtin_amdgcn_wmma_f32_16x16x32_f16(
            false, af, false, bf.v, (short)0, acc[mt], false, false);
      }
    }
  }

  #pragma unroll
  for (int mt = 0; mt < 4; ++mt) {
    v8h hv;
    #pragma unroll
    for (int v = 0; v < 8; ++v) hv[v] = (_Float16)fmaxf(acc[mt][v], 0.f);
    *(v8h*)(&stage[wv][n * C1 + mt * 16 + 8 * khalf]) = hv;
  }
  __syncthreads();

  long p0 = (long)b * HW + (long)(y0 + wv) * W_ + x0;
  const v8h* src = (const v8h*)(&stage[wv][lane * 32]);
  v8h* dst = (v8h*)(act2 + p0 * C1 + lane * 32);
  #pragma unroll
  for (int q = 0; q < 4; ++q) dst[q] = src[q];
}

// ---------------- per-(batch,segment) pixel counts ----------------
__global__ void counts_kernel(const int* __restrict__ segments,
                              float* __restrict__ counts) {
  int t = blockIdx.x * blockDim.x + threadIdx.x;
  if (t >= B_ * HW) return;
  int b = t / HW;
  atomicAdd(&counts[b * S_ + segments[t]], 1.0f);
}

// -------- conv3 (1x1, 64->768) + ReLU + fused segment scatter -------------
// kc2=0 half of w3F (48 KB) is staged block-wide in LDS by the Tensor Data
// Mover (TENSOR_LOAD_TO_LDS, 1-D D#, 8-byte elements), waited with
// s_wait_tensorcnt.  kc2=1 A-frags stream from global (L2-resident).
__global__ void __launch_bounds__(256)
conv3_wmma_seg_kernel(const _Float16* __restrict__ act2,
                      const _Float16* __restrict__ w3F,
                      const int* __restrict__ segments,
                      float* __restrict__ sums) {
  __shared__ _Float16 w3s[48 * 32 * 16];         // 49152 B, kc2=0 half

  // ---- TDM: DMA w3F[0 .. 49151] bytes into LDS (6144 x 8B elements) ----
  if (threadIdx.x < 32) {
    unsigned int ldsoff = (unsigned int)(unsigned long long)(&w3s[0]);
    unsigned long long ga = (unsigned long long)w3F;
    u32x4 g0;
    g0[0] = 1u;                                    // count=1, user mode
    g0[1] = ldsoff;                                // lds_addr
    g0[2] = (unsigned int)(ga & 0xffffffffu);      // global_addr[31:0]
    g0[3] = (unsigned int)((ga >> 32) & 0x1ffffffu) | (2u << 30); // addr[56:32], type=2
    i32x8 g1;
    g1[0] = (3 << 16);                             // data_size = 8B
    g1[1] = (int)(6144u << 16);                    // tensor_dim0[15:0] @ bits 63:48
    g1[2] = (int)(1u << 16);                       // tensor_dim0 hi=0, tensor_dim1=1
    g1[3] = (int)(6144u << 16);                    // tile_dim0 = 6144
    g1[4] = 0;                                     // tile_dim1/2 unused
    g1[5] = 6144;                                  // tensor_dim0_stride
    g1[6] = 0;
    g1[7] = 0;
    i32x4 gz = {0, 0, 0, 0};
#ifdef TDM_SIX_ARG
    i32x8 gz8 = {0, 0, 0, 0, 0, 0, 0, 0};
    __builtin_amdgcn_tensor_load_to_lds(g0, g1, gz, gz, gz8, 0);
#else
    __builtin_amdgcn_tensor_load_to_lds(g0, g1, gz, gz, 0);
#endif
    __builtin_amdgcn_s_wait_tensorcnt(0);
  }
  __syncthreads();

  int lane  = threadIdx.x & 31;
  int gw    = (blockIdx.x * blockDim.x + threadIdx.x) >> 5;
  int n     = lane & 15;
  int khalf = lane >> 4;
  long pix  = (long)gw * 16 + n;
  int  b    = (int)(pix / HW);

  const _Float16* src = act2 + pix * C1 + khalf * 8;
  Frag16 bf0, bf1;
  bf0.h[0] = *(const v8h*)(src);
  bf0.h[1] = *(const v8h*)(src + 16);
  bf1.h[0] = *(const v8h*)(src + 32);
  bf1.h[1] = *(const v8h*)(src + 48);

  int  sid     = segments[pix];
  long outbase = ((long)b * S_ + sid) * E_;

  const _Float16* w0 = w3s + lane * 16;                 // LDS (TDM-staged)
  const _Float16* w1 = w3F + 48 * 512 + (long)lane * 16; // global, kc2=1
  for (int mt = 0; mt < E_ / 16; ++mt) {
    v16h af0 = *(const v16h*)(w0 + mt * 512);
    v16h af1 = *(const v16h*)(w1 + (long)mt * 512);
    v8f acc = {};
    acc = __builtin_amdgcn_wmma_f32_16x16x32_f16(false, af0, false, bf0.v, (short)0, acc, false, false);
    acc = __builtin_amdgcn_wmma_f32_16x16x32_f16(false, af1, false, bf1.v, (short)0, acc, false, false);
    #pragma unroll
    for (int v = 0; v < 8; ++v) {
      int mo = mt * 16 + v + 8 * khalf;
      atomicAdd(&sums[outbase + mo], fmaxf(acc[v], 0.f));
    }
  }
}

// ---------------- embeddings = sums / max(counts,1) ----------------
__global__ void finalize_kernel(float* __restrict__ sums,
                                const float* __restrict__ counts) {
  int t = blockIdx.x * blockDim.x + threadIdx.x;
  if (t >= B_ * S_ * E_) return;
  float c = counts[t / E_];
  sums[t] = sums[t] / fmaxf(c, 1.0f);
}

// ---------------- pos MLP layer 1 ----------------
__global__ void hidden_kernel(const float* __restrict__ centroids,
                              const float* __restrict__ w1,
                              const float* __restrict__ b1,
                              float* __restrict__ hidden) {
  int t = blockIdx.x * blockDim.x + threadIdx.x;
  if (t >= B_ * S_ * HID) return;
  int j   = t % HID;
  int row = t / HID;
  const float inv = 1.0f / (float)(H_ - 1);
  float c0 = centroids[row * 2 + 0] * inv;
  float c1 = centroids[row * 2 + 1] * inv;
  hidden[t] = fmaxf(c0 * w1[j] + c1 * w1[HID + j] + b1[j], 0.f);
}

// ------- pos MLP layer 2: hidden[1024x384] @ w2[384x768] + b2 (f32 WMMA) --
__global__ void __launch_bounds__(256)
pos_wmma_kernel(const float* __restrict__ hidden,
                const float* __restrict__ w2,
                const float* __restrict__ b2,
                float* __restrict__ pos) {
  int lane  = threadIdx.x & 31;
  int gw    = (blockIdx.x * blockDim.x + threadIdx.x) >> 5;
  const int NT = E_ / 16;
  int mt = gw / NT, nt = gw % NT;
  int l15 = lane & 15, khalf = lane >> 4;
  int row = mt * 16 + l15;
  int n   = nt * 16 + l15;

  v8f acc = {};
  for (int kc = 0; kc < HID; kc += 4) {
    int kb = kc + 2 * khalf;
    v2f a = *(const v2f*)(hidden + row * HID + kb);
    v2f bvec;
    bvec[0] = w2[(kb + 0) * E_ + n];
    bvec[1] = w2[(kb + 1) * E_ + n];
    acc = __builtin_amdgcn_wmma_f32_16x16x4_f32(
        false, a, false, bvec, (short)0, acc, false, false);
  }
  float bias = b2[n];
  #pragma unroll
  for (int v = 0; v < 8; ++v) {
    int r = mt * 16 + v + 8 * khalf;
    pos[(long)r * E_ + n] = acc[v] + bias;
  }
}

extern "C" void kernel_launch(void* const* d_in, const int* in_sizes, int n_in,
                              void* d_out, int out_size, void* d_ws, size_t ws_size,
                              hipStream_t stream) {
  (void)in_sizes; (void)n_in; (void)out_size; (void)ws_size;
  const float* img       = (const float*)d_in[0];
  const int*   segments  = (const int*)d_in[1];
  const float* centroids = (const float*)d_in[2];
  const float* conv1_w   = (const float*)d_in[3];
  const float* conv2_w   = (const float*)d_in[4];
  const float* conv3_w   = (const float*)d_in[5];
  const float* pos_w1    = (const float*)d_in[6];
  const float* pos_b1    = (const float*)d_in[7];
  const float* pos_w2    = (const float*)d_in[8];
  const float* pos_b2    = (const float*)d_in[9];

  float* emb = (float*)d_out;                         // [B,S,E]
  float* pos = (float*)d_out + (size_t)B_ * S_ * E_;  // [B,S,E]

  char* ws = (char*)d_ws;
  _Float16* act1 = (_Float16*)(ws);                       // 25,690,112 B (NHWC)
  _Float16* act2 = (_Float16*)(ws + 25690112);            // 25,690,112 B (NHWC)
  _Float16* w2F  = (_Float16*)(ws + 51380224);            //     73,728 B
  _Float16* w3F  = (_Float16*)(ws + 51453952);            //     98,304 B
  float*    hid  = (float*)   (ws + 51552256);            //  1,572,864 B
  float*    cnt  = (float*)   (ws + 53125120);            //      4,096 B

  hipMemsetAsync(emb, 0, (size_t)B_ * S_ * E_ * sizeof(float), stream);
  hipMemsetAsync(cnt, 0, B_ * S_ * sizeof(float), stream);

  conv1_kernel<<<(B_ * C1 * HW + 255) / 256, 256, 0, stream>>>(img, conv1_w, act1);
  cvt_kernel<<<(36864 + 49152 + 255) / 256, 256, 0, stream>>>(conv2_w, conv3_w, w2F, w3F);

  int conv2_blocks = B_ * (H_ / 8) * (W_ / 16);        // 1568
  conv2_wmma_kernel<<<conv2_blocks, 256, 0, stream>>>(act1, w2F, act2);
  counts_kernel<<<(B_ * HW + 255) / 256, 256, 0, stream>>>(segments, cnt);

  int conv3_blocks = (B_ * HW / 16) * 32 / 256;        // 1568
  conv3_wmma_seg_kernel<<<conv3_blocks, 256, 0, stream>>>(act2, w3F, segments, emb);
  finalize_kernel<<<(B_ * S_ * E_ + 255) / 256, 256, 0, stream>>>(emb, cnt);

  hidden_kernel<<<(B_ * S_ * HID + 255) / 256, 256, 0, stream>>>(centroids, pos_w1, pos_b1, hid);
  pos_wmma_kernel<<<(64 * (E_ / 16) * 32) / 256, 256, 0, stream>>>(hid, pos_w2, pos_b2, pos);
}
